// _ProposalLayer_35811437314321
// MI455X (gfx1250) — compile-verified, matrix operand order checked by reading, not run on hardware
//
#include <hip/hip_runtime.h>
#include <hip/hip_bf16.h>
#include <math.h>

typedef int v8i __attribute__((ext_vector_type(8)));

#define BB 4
#define AA 9
#define HH 50
#define WW 76
#define NN (AA * HH * WW)          // 34200 proposals per image
#define NPAD 34240                 // NN rounded up to multiple of 64 (535*64)
#define PRE 6000                   // PRE_NMS_TOPN
#define POST 300                   // POST_NMS_TOPN
#define NMS_TH 0.7f
#define MASK_W ((PRE + 31) / 32)   // 188 u32 words per mask row

#define AS1 __attribute__((address_space(1)))
#define AS3 __attribute__((address_space(3)))

__device__ __forceinline__ void wait_async0() {
#if __has_builtin(__builtin_amdgcn_s_wait_asynccnt)
  __builtin_amdgcn_s_wait_asynccnt(0);
#else
  asm volatile("s_wait_asynccnt 0x0" ::: "memory");
#endif
}

// One B128 async copy: LDS[dst] = global[src], tracked on ASYNCcnt.
__device__ __forceinline__ void async_copy_b128(const float* src, float* dst_lds) {
  asm volatile("global_load_async_to_lds_b128 %0, %1, off"
               :
               : "v"((AS3 float*)dst_lds), "v"((AS1 const float*)src)
               : "memory");
}

// ---------------------------------------------------------------------------
// Stage 1: anchor generation + bbox decode + clip.  boxes[(b*NN+i)*4],
// scs[b*NPAD+i] with pad scores = -1.0f so async staging needs no bounds select.
// i = (h*W + w)*A + a  (matches reference transpose(0,2,3,1).reshape ordering)
// ---------------------------------------------------------------------------
__global__ void decode_kernel(const float* __restrict__ scores_in,
                              const float* __restrict__ deltas_in,
                              const float* __restrict__ im_info,
                              float* __restrict__ boxes,
                              float* __restrict__ scs) {
  int idx = blockIdx.x * blockDim.x + threadIdx.x;
  if (idx >= BB * NPAD) return;
  int b = idx / NPAD;
  int i = idx - b * NPAD;
  if (i >= NN) { scs[idx] = -1.0f; return; }   // pad tail
  int a = i % AA;
  int s = i / AA;
  int w = s % WW;
  int h = s / WW;

  // generate_anchors(): ratio index = a/3, scale index = a%3; center stays 7.5
  const float ratios[3] = {0.5f, 1.0f, 2.0f};
  const float scales[3] = {8.0f, 16.0f, 32.0f};
  float ratio = ratios[a / 3];
  float scale = scales[a % 3];
  float wsr = roundf(sqrtf(256.0f / ratio));
  float hsr = roundf(wsr * ratio);
  float aw = wsr * scale;
  float ah = hsr * scale;
  float sx = (float)(w * 16);    // FEAT_STRIDE
  float sy = (float)(h * 16);
  float x1 = sx + 7.5f - 0.5f * (aw - 1.0f);
  float y1 = sy + 7.5f - 0.5f * (ah - 1.0f);
  float x2 = sx + 7.5f + 0.5f * (aw - 1.0f);
  float y2 = sy + 7.5f + 0.5f * (ah - 1.0f);
  float wa  = x2 - x1 + 1.0f;
  float ha  = y2 - y1 + 1.0f;
  float cxa = x1 + 0.5f * wa;
  float cya = y1 + 0.5f * ha;

  size_t plane = (size_t)HH * WW;
  size_t base  = (((size_t)b * (4 * AA) + 4 * a) * HH + h) * WW + w;
  float dx = deltas_in[base];
  float dy = deltas_in[base + plane];
  float dw = deltas_in[base + 2 * plane];
  float dh = deltas_in[base + 3 * plane];

  float pcx = dx * wa + cxa;
  float pcy = dy * ha + cya;
  float pw  = expf(dw) * wa;
  float ph  = expf(dh) * ha;
  float bx1 = pcx - 0.5f * pw;
  float by1 = pcy - 0.5f * ph;
  float bx2 = pcx + 0.5f * pw;
  float by2 = pcy + 0.5f * ph;

  float imh = im_info[b * 3 + 0];
  float imw = im_info[b * 3 + 1];
  bx1 = fminf(fmaxf(bx1, 0.0f), imw - 1.0f);
  bx2 = fminf(fmaxf(bx2, 0.0f), imw - 1.0f);
  by1 = fminf(fmaxf(by1, 0.0f), imh - 1.0f);
  by2 = fminf(fmaxf(by2, 0.0f), imh - 1.0f);

  float sc = scores_in[(((size_t)b * (2 * AA) + AA + a) * HH + h) * WW + w];
  size_t o = ((size_t)b * NN + i) * 4;
  boxes[o + 0] = bx1;
  boxes[o + 1] = by1;
  boxes[o + 2] = bx2;
  boxes[o + 3] = by2;
  scs[idx] = sc;
}

// ---------------------------------------------------------------------------
// Stage 2: exact descending rank of every score (tie-break: lower index wins),
// computed with V_WMMA_I32_16X16X64_IU8.  rank_i = #{j : key_j > key_i}.
// A = 16x64 u8 indicator matrix (lane L & L+16 jointly hold row L%16; with an
// all-ones B the K placement within the pair is irrelevant: C = row-sums).
// C accumulates ranks across the whole j sweep.  rank < 6000 -> sortedIdx[rank].
// j chunk of 64 scores staged into LDS once per block via async-to-LDS B128
// (16 lanes x 16B), shared by all 8 waves; scores array is pre-padded.
// ---------------------------------------------------------------------------
__global__ void rank_topk_kernel(const float* __restrict__ scs,
                                 int* __restrict__ sidx) {
  int b    = blockIdx.y;
  int tid  = threadIdx.x;
  int lane = tid & 31;
  int wv   = tid >> 5;
  int tile = blockIdx.x * 8 + wv;
  int m    = lane & 15;                 // A-matrix row held by this lane
  int gi   = tile * 16 + m;             // global proposal index of this row
  float si = (gi < NN) ? scs[(size_t)b * NPAD + gi] : -1.0f;  // scores in [0,1)

  __shared__ __align__(16) float shs[64];

  v8i bones, c;
#pragma unroll
  for (int v = 0; v < 8; ++v) { bones[v] = 0x01010101; c[v] = 0; }

  int half = (lane < 16) ? 0 : 32;      // each half-wave covers 32 of 64 j's

  for (int j0 = 0; j0 < NPAD; j0 += 64) {
    __syncthreads();                    // previous chunk fully consumed
    if (tid < 16)                       // 16 x B128 = 64 floats, async pipe
      async_copy_b128(scs + (size_t)b * NPAD + j0 + tid * 4, &shs[tid * 4]);
    wait_async0();
    __syncthreads();

    v8i av;
#pragma unroll
    for (int v = 0; v < 8; ++v) {
      unsigned wbits = 0u;
#pragma unroll
      for (int t = 0; t < 4; ++t) {
        int off  = half + v * 4 + t;
        float sj = shs[off];
        int j    = j0 + off;
        bool g   = (sj > si) || ((sj == si) && (j < gi));
        wbits |= g ? (1u << (8 * t)) : 0u;            // u8 indicator 0/1
      }
      av[v] = (int)wbits;
    }
    // D(i32 16x16) = A(16x64 u8) * ones(64x16 u8) + C  -> per-row counts
    c = __builtin_amdgcn_wmma_i32_16x16x64_iu8(false, av, false, bones, c,
                                               false, false);
  }

  // C layout: lane<16 VGPR v holds M=v; lane>=16 VGPR v holds M=8+v (any N col)
  if (lane == 0 || lane == 16) {
    int mb = (lane == 0) ? 0 : 8;
#pragma unroll
    for (int v = 0; v < 8; ++v) {
      int row = tile * 16 + mb + v;
      if (row < NN) {
        int r = c[v];
        if (r < PRE) sidx[b * PRE + r] = row;   // ranks are a bijection
      }
    }
  }
}

// ---------------------------------------------------------------------------
// Stage 3: gather sorted boxes (float4 loads/stores, 16B aligned)
// ---------------------------------------------------------------------------
__global__ void gather_kernel(const float* __restrict__ boxes,
                              const int* __restrict__ sidx,
                              float* __restrict__ sbox) {
  int t = blockIdx.x * blockDim.x + threadIdx.x;
  if (t >= BB * PRE) return;
  int b = t / PRE;
  int src = sidx[t];
  ((float4*)sbox)[t] = ((const float4*)boxes)[(size_t)b * NN + src];
}

// ---------------------------------------------------------------------------
// Stage 4: pairwise suppression bitmask.  bit(i,j) = (j>i) & (IoU>0.7),
// legacy +1 box widths, strict >.  One wave per 32x32 tile; the 32 column
// boxes are staged into LDS with async-to-LDS B128 (one per lane).  The tail
// block over-reads <=256B into the adjacent allocated mask region; those
// columns are discarded by the j<PRE bit condition.
// ---------------------------------------------------------------------------
__global__ void iou_mask_kernel(const float* __restrict__ sbox,
                                unsigned* __restrict__ mask) {
  int b = blockIdx.z;
  int rowBase = blockIdx.y * 32;
  int colBase = blockIdx.x * 32;
  int lane = threadIdx.x;

  __shared__ __align__(16) float cbox[32][4];
  async_copy_b128(sbox + ((size_t)b * PRE + (colBase + lane)) * 4,
                  &cbox[lane][0]);
  wait_async0();
  __syncthreads();

  int row = rowBase + lane;
  if (row >= PRE) return;
  float4 rb = ((const float4*)sbox)[(size_t)b * PRE + row];
  float ra = (rb.z - rb.x + 1.0f) * (rb.w - rb.y + 1.0f);

  unsigned bits = 0u;
  for (int t = 0; t < 32; ++t) {
    int j = colBase + t;
    float jx1 = cbox[t][0], jy1 = cbox[t][1];
    float jx2 = cbox[t][2], jy2 = cbox[t][3];
    float ja = (jx2 - jx1 + 1.0f) * (jy2 - jy1 + 1.0f);
    float iw = fminf(rb.z, jx2) - fmaxf(rb.x, jx1) + 1.0f;
    float ih = fminf(rb.w, jy2) - fmaxf(rb.y, jy1) + 1.0f;
    iw = fmaxf(iw, 0.0f);
    ih = fmaxf(ih, 0.0f);
    float inter = iw * ih;
    float iou = inter / (ra + ja - inter);
    if ((j < PRE) && (j > row) && (iou > NMS_TH)) bits |= (1u << t);
  }
  mask[((size_t)b * PRE + row) * MASK_W + blockIdx.x] = bits;
}

// ---------------------------------------------------------------------------
// Stage 5: sequential greedy scan per image (one block each).  188-word
// "removed" bitmask lives in LDS; kept counter tracked redundantly per thread
// (decision is block-uniform).  Output zero-init + batch index on all rows.
// ---------------------------------------------------------------------------
__global__ void nms_scan_kernel(const unsigned* __restrict__ mask,
                                const float* __restrict__ sbox,
                                float* __restrict__ out) {
  int b = blockIdx.x;
  int tid = threadIdx.x;
  __shared__ unsigned removed[MASK_W];
  for (int t = tid; t < MASK_W; t += blockDim.x) removed[t] = 0u;
  for (int t = tid; t < POST * 5; t += blockDim.x)
    out[b * POST * 5 + t] = ((t % 5) == 0) ? (float)b : 0.0f;
  __syncthreads();

  int kept = 0;   // block-uniform by construction
  for (int i = 0; i < PRE; ++i) {
    // bit i is settled: rows processed this iteration only set bits > i
    bool rem = (removed[i >> 5] >> (i & 31)) & 1u;
    if (tid == 0 && (i + 1) < PRE)
      __builtin_prefetch(&mask[((size_t)b * PRE + i + 1) * MASK_W], 0, 0);
    if (!rem) {
      if (kept < POST) {
        if (tid < 4)
          out[((size_t)b * POST + kept) * 5 + 1 + tid] =
              sbox[((size_t)b * PRE + i) * 4 + tid];
      }
      const unsigned* mrow = &mask[((size_t)b * PRE + i) * MASK_W];
      for (int t = tid; t < MASK_W; t += blockDim.x) {
        unsigned mv = mrow[t];
        if (mv) atomicOr(&removed[t], mv);
      }
      ++kept;
    }
    __syncthreads();
  }
}

// ---------------------------------------------------------------------------
extern "C" void kernel_launch(void* const* d_in, const int* in_sizes, int n_in,
                              void* d_out, int out_size, void* d_ws, size_t ws_size,
                              hipStream_t stream) {
  (void)in_sizes; (void)n_in; (void)out_size; (void)ws_size;
  const float* scores = (const float*)d_in[0];
  const float* deltas = (const float*)d_in[1];
  const float* iminfo = (const float*)d_in[2];
  float* out = (float*)d_out;

  // workspace carve-up (all 16B-aligned)
  float*    boxes = (float*)d_ws;                              // BB*NN*4
  float*    scs   = boxes + (size_t)BB * NN * 4;               // BB*NPAD
  int*      sidx  = (int*)(scs + (size_t)BB * NPAD);           // BB*PRE
  float*    sbox  = (float*)(sidx + BB * PRE);                 // BB*PRE*4
  unsigned* mask  = (unsigned*)(sbox + (size_t)BB * PRE * 4);  // BB*PRE*MASK_W

  int total = BB * NPAD;
  decode_kernel<<<(total + 255) / 256, 256, 0, stream>>>(scores, deltas, iminfo,
                                                         boxes, scs);

  int tilesPerImg  = (NN + 15) / 16;        // 2138
  int blocksPerImg = (tilesPerImg + 7) / 8; // 268 (8 waves/block)
  rank_topk_kernel<<<dim3(blocksPerImg, BB), 256, 0, stream>>>(scs, sidx);

  gather_kernel<<<(BB * PRE + 255) / 256, 256, 0, stream>>>(boxes, sidx, sbox);

  iou_mask_kernel<<<dim3(MASK_W, MASK_W, BB), 32, 0, stream>>>(sbox, mask);

  nms_scan_kernel<<<BB, 256, 0, stream>>>(mask, sbox, out);
}
